// HybridLFISTA_18468359373310
// MI455X (gfx1250) — matrix operand order repaired and checked by reading, not compiled, not run on hardware
//
#include <hip/hip_runtime.h>
#include <math.h>

// ---------------------------------------------------------------------------
// HybridLFISTA on gfx1250 (MI455X): all GEMMs via v_wmma_f32_16x16x32_bf16.
// Activations are converted to bf16 once (at production); GEMM staging is a
// pure bf16 copy (global_load_b128 -> ds_store), double-buffered, one barrier
// per 32-wide K-slab.
// ---------------------------------------------------------------------------

typedef __attribute__((ext_vector_type(16))) __bf16 v16bf;
typedef __attribute__((ext_vector_type(8)))  float  v8f;

union FragU { v16bf v; unsigned int u[8]; };
union AccU  { v8f  v; float f[8]; };

__device__ __forceinline__ unsigned short f2bf(float f) {
    unsigned int u = __float_as_uint(f);
    unsigned int r = u + 0x7FFFu + ((u >> 16) & 1u);   // round-to-nearest-even
    return (unsigned short)(r >> 16);
}

__device__ __forceinline__ float clip_exp(float v) {
    return fminf(fmaxf(__expf(v), 1e-8f), 10.0f);
}

// ---- elementwise helpers ---------------------------------------------------

__global__ void k_convert_bf16(const float* __restrict__ in,
                               unsigned short* __restrict__ out, int n) {
    int i = blockIdx.x * blockDim.x + threadIdx.x;
    if (i < n) out[i] = f2bf(in[i]);
}

// in: fp32 [R][C]  ->  out: bf16 [C][R]
__global__ void k_transpose_bf16(const float* __restrict__ in,
                                 unsigned short* __restrict__ out, int R, int C) {
    int i = blockIdx.x * blockDim.x + threadIdx.x;
    if (i < R * C) {
        int r = i / C, c = i - r * C;
        out[(size_t)c * R + r] = f2bf(in[i]);
    }
}

// v = cur + beta*(cur - prev); dual fp32 + bf16 outputs
__global__ void k_momentum(const float* __restrict__ cur,
                           const float* __restrict__ prev,
                           float* __restrict__ v,
                           unsigned short* __restrict__ vb,
                           float beta, int n) {
    int i = blockIdx.x * blockDim.x + threadIdx.x;
    if (i < n) {
        float a = cur[i];
        float r = a + beta * (a - prev[i]);
        v[i]  = r;
        vb[i] = f2bf(r);
    }
}

// ---- tiled WMMA GEMM: epi( Abf[M][K] * Bw[N][K]^T ) ------------------------
// Block tile 128x128, K-slab 32, LDS double-buffered (1 barrier per slab).
// 256 threads = 8 waves in a 4(M) x 2(N) grid; each wave owns a 32x64 patch
// = 2x4 WMMA 16x16 tiles. Both operands arrive as bf16 [rows][K]; staging is
// a straight 128-bit copy.
//
// EPI: 1 bias+GELU(exact) | 2 bias | 3 aux-acc | 4 acc-aux | 5 FISTA soft-thr
//      6 aux+acc
// Cf (fp32) and Cbf (bf16 shadow) outputs are each optional (nullable).

template <int EPI>
__global__ __launch_bounds__(256)
void k_gemm_bf16(const unsigned short* __restrict__ Abf,
                 const unsigned short* __restrict__ Bw,
                 float* __restrict__ Cf,
                 unsigned short* __restrict__ Cbf,
                 int N, int Kd,
                 const float* __restrict__ aux0,
                 const float* __restrict__ bias,
                 const float* __restrict__ p_eta,
                 const float* __restrict__ p_tau, int kidx) {
    // 128 rows x 32 bf16 (16 dwords) padded to 17 dwords/row: lane-to-lane
    // stride 17 banks, coprime with 64 -> conflict-free fragment gathers.
    __shared__ unsigned int lds_a[2][128 * 17];
    __shared__ unsigned int lds_b[2][128 * 17];

    const int tid  = threadIdx.x;
    const int lane = tid & 31;
    const int wave = tid >> 5;
    const int wm   = wave & 3;     // 0..3  -> 32-row band
    const int wn   = wave >> 2;    // 0..1  -> 64-col band
    const int m0   = blockIdx.y * 128;
    const int n0   = blockIdx.x * 128;

    AccU acc[2][4];
#pragma unroll
    for (int i = 0; i < 2; ++i)
#pragma unroll
        for (int t = 0; t < 4; ++t)
#pragma unroll
            for (int e = 0; e < 8; ++e) acc[i][t].f[e] = 0.0f;

    const int srow  = tid >> 2;        // 0..63: staging row per pass
    const int scol4 = (tid & 3) * 4;   // dword column (uint4 granule)

    const int khalf = lane >> 4;
    const int l15   = lane & 15;

    auto stage = [&](int kt, int buf) {
#pragma unroll
        for (int i = 0; i < 2; ++i) {
            int r = srow + 64 * i;
            const uint4 qa = *reinterpret_cast<const uint4*>(
                &Abf[(size_t)(m0 + r) * Kd + kt + scol4 * 2]);
            lds_a[buf][r * 17 + scol4 + 0] = qa.x;
            lds_a[buf][r * 17 + scol4 + 1] = qa.y;
            lds_a[buf][r * 17 + scol4 + 2] = qa.z;
            lds_a[buf][r * 17 + scol4 + 3] = qa.w;
            const uint4 qb = *reinterpret_cast<const uint4*>(
                &Bw[(size_t)(n0 + r) * Kd + kt + scol4 * 2]);
            lds_b[buf][r * 17 + scol4 + 0] = qb.x;
            lds_b[buf][r * 17 + scol4 + 1] = qb.y;
            lds_b[buf][r * 17 + scol4 + 2] = qb.z;
            lds_b[buf][r * 17 + scol4 + 3] = qb.w;
        }
    };

    stage(0, 0);
    __syncthreads();

    int cur = 0;
    for (int kt = 0; kt < Kd; kt += 32) {
        // Stage next slab into the other buffer while this one is consumed.
        if (kt + 32 < Kd) stage(kt + 32, cur ^ 1);
        if (kt + 64 < Kd) {  // global_prefetch_b8 of slab after next
            __builtin_prefetch(&Abf[(size_t)(m0 + srow) * Kd + kt + 64 + scol4 * 2], 0, 3);
            __builtin_prefetch(&Bw[(size_t)(n0 + srow) * Kd + kt + 64 + scol4 * 2], 0, 3);
        }

        // ---- gather fragments per ISA VGPR layouts ----
        FragU af[2], bf[4];
#pragma unroll
        for (int i = 0; i < 2; ++i) {
            int row = wm * 32 + i * 16 + l15;            // M index
#pragma unroll
            for (int j = 0; j < 8; ++j) {
                // A 16x32 bf16: j<4 -> K=khalf*8+2j ; j>=4 -> K=16+khalf*8+2(j-4)
                int pidx = (j < 4) ? (khalf * 4 + j) : (4 + khalf * 4 + j);
                af[i].u[j] = lds_a[cur][row * 17 + pidx];
            }
        }
#pragma unroll
        for (int t = 0; t < 4; ++t) {
            int row = wn * 64 + t * 16 + l15;            // N index
#pragma unroll
            for (int j = 0; j < 8; ++j) {
                // B 32x16 bf16: j<4 -> K=khalf*8+2j ; j>=4 -> K=16+khalf*8+2(j-4)
                bf[t].u[j] = lds_b[cur][row * 17 + ((j < 4) ? (khalf * 4 + j) : (4 + khalf * 4 + j))];
            }
        }

#pragma unroll
        for (int i = 0; i < 2; ++i)
#pragma unroll
            for (int t = 0; t < 4; ++t)
                acc[i][t].v = __builtin_amdgcn_wmma_f32_16x16x32_bf16(
                    false, af[i].v, false, bf[t].v,
                    (short)0, acc[i][t].v, false, false);

        __syncthreads();    // single barrier per slab (double-buffered)
        cur ^= 1;
    }

    // ---- fused epilogue ----
    float eta = 0.0f, thr = 0.0f;
    if constexpr (EPI == 5) {
        eta = clip_exp(p_eta[kidx]);
        thr = eta * clip_exp(p_tau[kidx]);
    }
#pragma unroll
    for (int i = 0; i < 2; ++i) {
#pragma unroll
        for (int t = 0; t < 4; ++t) {
#pragma unroll
            for (int e = 0; e < 8; ++e) {
                int mg = m0 + wm * 32 + i * 16 + e + 8 * khalf;
                int ng = n0 + wn * 64 + t * 16 + l15;
                size_t o = (size_t)mg * N + ng;
                float a = acc[i][t].f[e];
                float r;
                if constexpr (EPI == 1) {
                    float s = a + bias[ng];
                    r = 0.5f * s * (1.0f + erff(s * 0.7071067811865476f));
                } else if constexpr (EPI == 2) {
                    r = a + bias[ng];
                } else if constexpr (EPI == 3) {
                    r = aux0[o] - a;                   // z = b - ybg@M^T
                } else if constexpr (EPI == 4) {
                    r = a - aux0[o];                   // r = v@A^T - z
                } else if constexpr (EPI == 5) {
                    float u = aux0[o] - eta * a;       // v - eta*grad
                    float m = fabsf(u) - thr;
                    r = (m > 0.0f) ? copysignf(m, u) : 0.0f;
                } else if constexpr (EPI == 6) {
                    r = aux0[o] + a;                   // y_hat = y_bg + alpha@Psi^T
                } else {
                    r = a;
                }
                if (Cf)  Cf[o]  = r;
                if (Cbf) Cbf[o] = f2bf(r);
            }
        }
    }
}

// ---------------------------------------------------------------------------

extern "C" void kernel_launch(void* const* d_in, const int* in_sizes, int n_in,
                              void* d_out, int out_size, void* d_ws, size_t ws_size,
                              hipStream_t stream) {
    (void)in_sizes; (void)n_in; (void)out_size; (void)ws_size;
    enum { Bb = 8192, P = 512, H1 = 1024, H2 = 512, NOUT = 1024,
           MM = 512, NCODE = 2048, KSTEPS = 16 };

    const float* x       = (const float*)d_in[0];
    const float* bmeas   = (const float*)d_in[1];
    const float* W1      = (const float*)d_in[2];
    const float* b1      = (const float*)d_in[3];
    const float* W2      = (const float*)d_in[4];
    const float* b2      = (const float*)d_in[5];
    const float* W3      = (const float*)d_in[6];
    const float* b3      = (const float*)d_in[7];
    const float* Amat    = (const float*)d_in[8];
    const float* log_eta = (const float*)d_in[9];
    const float* log_tau = (const float*)d_in[10];
    const float* Psi     = (const float*)d_in[11];
    const float* Mmat    = (const float*)d_in[12];

    float* out       = (float*)d_out;
    float* out_ybg   = out;                                   // [B, NOUT]
    float* out_z     = out_ybg + (size_t)Bb * NOUT;           // [B, MM]
    float* out_alpha = out_z + (size_t)Bb * MM;               // [B, NCODE]
    float* out_yhat  = out_alpha + (size_t)Bb * NCODE;        // [B, NOUT]

    // ---- workspace carve ----
    char* wp = (char*)d_ws;
    auto take = [&](size_t bytes) -> void* {
        void* r = wp; wp += (bytes + 255) & ~(size_t)255; return r;
    };
    // fp32 FISTA state
    float* a0   = (float*)take((size_t)Bb * NCODE * 4);
    float* a1   = (float*)take((size_t)Bb * NCODE * 4);
    float* vbuf = (float*)take((size_t)Bb * NCODE * 4);
    // bf16 activation shadows (WMMA A-operands)
    unsigned short* xbf     = (unsigned short*)take((size_t)Bb * P * 2);
    unsigned short* h1bf    = (unsigned short*)take((size_t)Bb * H1 * 2);
    unsigned short* h2bf    = (unsigned short*)take((size_t)Bb * H2 * 2);
    unsigned short* ybgbf   = (unsigned short*)take((size_t)Bb * NOUT * 2);
    unsigned short* vbf     = (unsigned short*)take((size_t)Bb * NCODE * 2);
    unsigned short* rbf     = (unsigned short*)take((size_t)Bb * MM * 2);
    unsigned short* alphabf = (unsigned short*)take((size_t)Bb * NCODE * 2);
    // bf16 weights in [N][K] form
    unsigned short* W1t   = (unsigned short*)take((size_t)P * H1 * 2);
    unsigned short* W2t   = (unsigned short*)take((size_t)H1 * H2 * 2);
    unsigned short* W3t   = (unsigned short*)take((size_t)H2 * NOUT * 2);
    unsigned short* Abf   = (unsigned short*)take((size_t)MM * NCODE * 2);
    unsigned short* At    = (unsigned short*)take((size_t)MM * NCODE * 2);
    unsigned short* Mbf   = (unsigned short*)take((size_t)MM * NOUT * 2);
    unsigned short* Psibf = (unsigned short*)take((size_t)NOUT * NCODE * 2);

    const int TB = 256;
    auto blocks = [](size_t n) { return (unsigned)((n + 255) / 256); };

    // ---- one-time conversions ----
    k_convert_bf16<<<blocks((size_t)Bb * P), TB, 0, stream>>>(x, xbf, Bb * P);
    k_transpose_bf16<<<blocks((size_t)P * H1), TB, 0, stream>>>(W1, W1t, P, H1);
    k_transpose_bf16<<<blocks((size_t)H1 * H2), TB, 0, stream>>>(W2, W2t, H1, H2);
    k_transpose_bf16<<<blocks((size_t)H2 * NOUT), TB, 0, stream>>>(W3, W3t, H2, NOUT);
    k_transpose_bf16<<<blocks((size_t)MM * NCODE), TB, 0, stream>>>(Amat, At, MM, NCODE);
    k_convert_bf16<<<blocks((size_t)MM * NCODE), TB, 0, stream>>>(Amat, Abf, MM * NCODE);
    k_convert_bf16<<<blocks((size_t)MM * NOUT), TB, 0, stream>>>(Mmat, Mbf, MM * NOUT);
    k_convert_bf16<<<blocks((size_t)NOUT * NCODE), TB, 0, stream>>>(Psi, Psibf, NOUT * NCODE);

    // ---- MLP background (hidden activations kept bf16-only) ----
    dim3 g1(H1 / 128, Bb / 128);
    k_gemm_bf16<1><<<g1, TB, 0, stream>>>(xbf, W1t, nullptr, h1bf, H1, P,
                                          nullptr, b1, nullptr, nullptr, 0);
    dim3 g2(H2 / 128, Bb / 128);
    k_gemm_bf16<1><<<g2, TB, 0, stream>>>(h1bf, W2t, nullptr, h2bf, H2, H1,
                                          nullptr, b2, nullptr, nullptr, 0);
    dim3 g3(NOUT / 128, Bb / 128);
    k_gemm_bf16<2><<<g3, TB, 0, stream>>>(h2bf, W3t, out_ybg, ybgbf, NOUT, H2,
                                          nullptr, b3, nullptr, nullptr, 0);
    // z = b - y_bg @ M^T
    dim3 gz(MM / 128, Bb / 128);
    k_gemm_bf16<3><<<gz, TB, 0, stream>>>(ybgbf, Mbf, out_z, nullptr, MM, NOUT,
                                          bmeas, nullptr, nullptr, nullptr, 0);

    // ---- FISTA ----
    hipMemsetAsync(a0, 0, (size_t)Bb * NCODE * 4, stream);
    hipMemsetAsync(a1, 0, (size_t)Bb * NCODE * 4, stream);

    double t_prev = 1.0;
    float* cur = a0;
    float* prv = a1;
    const int nAlpha = Bb * NCODE;
    dim3 gr(MM / 128, Bb / 128);       // r GEMM: [B,2048]x[2048,512]
    dim3 ga(NCODE / 128, Bb / 128);    // grad GEMM: [B,512]x[512,2048]
    for (int k = 0; k < KSTEPS; ++k) {
        float beta = 0.0f;
        if (k > 0) {
            double t_new = 0.5 * (1.0 + sqrt(1.0 + 4.0 * t_prev * t_prev));
            beta = (float)((t_prev - 1.0) / t_new);
            t_prev = t_new;
        }
        k_momentum<<<blocks((size_t)nAlpha), TB, 0, stream>>>(cur, prv, vbuf, vbf,
                                                              beta, nAlpha);
        // r = v @ A^T - z   (bf16-only output; feeds next GEMM's A operand)
        k_gemm_bf16<4><<<gr, TB, 0, stream>>>(vbf, Abf, nullptr, rbf, MM, NCODE,
                                              out_z, nullptr, nullptr, nullptr, 0);
        // alpha_new = soft(v - eta*(r @ A), eta*tau)
        float* dst = (k == KSTEPS - 1) ? out_alpha : prv;
        unsigned short* dstbf = (k == KSTEPS - 1) ? alphabf : nullptr;
        k_gemm_bf16<5><<<ga, TB, 0, stream>>>(rbf, At, dst, dstbf, NCODE, MM,
                                              vbuf, nullptr, log_eta, log_tau, k);
        prv = cur;
        cur = dst;
    }

    // y_hat = y_bg + alpha @ Psi^T
    dim3 gy(NOUT / 128, Bb / 128);
    k_gemm_bf16<6><<<gy, TB, 0, stream>>>(alphabf, Psibf, out_yhat, nullptr, NOUT, NCODE,
                                          out_ybg, nullptr, nullptr, nullptr, 0);
}
